// VectorQuantizer_7275674599938
// MI455X (gfx1250) — matrix operand order repaired and checked by reading, not compiled, not run on hardware
//
#include <hip/hip_runtime.h>
#include <hip/hip_bf16.h>

typedef __attribute__((ext_vector_type(16))) _Float16 v16h;
typedef __attribute__((ext_vector_type(8)))  float    v8f;

#define NUM_EMB   1024
#define EMB_DIM   64
#define N_ROWS    32768          // 16 * 2048
#define COMMIT    0.25f

// ---------------- workspace layout (bytes) ----------------
// [0,      4096)   : unsigned hist[1024]
// [4096,   4100)   : float    sse
// [8192,   139264) : int      idx[N_ROWS]
// [139264, 143360) : float    enorm[1024]          (exact f32 ||e_n||^2)
// [143360, 405504) : v16h     bfrag[64*4*32]       (pre-swizzled codebook B fragments,
//                                                   per tile: b0h,b0l,b1h,b1l x 32 lanes)
#define WS_SSE_OFF    4096
#define WS_IDX_OFF    8192
#define WS_EN_OFF     139264
#define WS_BF_OFF     143360

__device__ __forceinline__ v8f wmma_f16(v16h a, v16h b, v8f c) {
    // D = A(16x32 f16) x B(32x16 f16) + C(16x16 f32)
    return __builtin_amdgcn_wmma_f32_16x16x32_f16(
        /*neg_a=*/false, a, /*neg_b=*/false, b,
        /*c_mod=*/(short)0, c, /*reuse_a=*/false, /*reuse_b=*/false);
}

// A fragment (16x32 f16, MxK): lane = M%16; half 0 holds K {kc..kc+7, kc+16..kc+23},
// half 1 holds K {kc+8..kc+15, kc+24..kc+31}. Split each f32 into hi+lo f16.
__device__ __forceinline__ void load_a_frag(const float* __restrict__ rowPtr, int kc, int half,
                                            v16h& hi, v16h& lo) {
    const float4* p = (const float4*)rowPtr;
    const int b0 = (kc + half * 8) >> 2;
    float4 x0 = p[b0], x1 = p[b0 + 1], x2 = p[b0 + 4], x3 = p[b0 + 5];
    float f[16] = {x0.x,x0.y,x0.z,x0.w, x1.x,x1.y,x1.z,x1.w,
                   x2.x,x2.y,x2.z,x2.w, x3.x,x3.y,x3.z,x3.w};
#pragma unroll
    for (int i = 0; i < 16; ++i) {
        _Float16 h = (_Float16)f[i];
        hi[i] = h;
        lo[i] = (_Float16)(f[i] - (float)h);
    }
}

__global__ void vq_init(unsigned* __restrict__ hist, float* __restrict__ sse) {
    int t = threadIdx.x;
    if (t < NUM_EMB) hist[t] = 0u;
    if (t == 0) *sse = 0.f;
}

// Pass 0: pre-swizzle codebook into B-fragment layout (hi/lo f16) + exact enorm.
// B fragment (32x16 f16, KxN): lane col N = lane%16; half 0 holds K {kc..kc+7, kc+16..kc+23},
// half 1 holds K {kc+8..kc+15, kc+24..kc+31} (matches A fragment order).
// One thread per (tile nt, lane): 64*32 = 2048 threads.
__global__ void __launch_bounds__(256)
vq_prep(const float* __restrict__ cb, v16h* __restrict__ bw, float* __restrict__ en) {
    const int t    = blockIdx.x * blockDim.x + threadIdx.x;
    const int nt   = t >> 5;
    const int lane = t & 31;
    const int h    = lane >> 4;
    const int lr   = lane & 15;
    const float* crow = cb + (size_t)(nt * 16 + lr) * EMB_DIM;

#pragma unroll
    for (int kc = 0; kc < 2; ++kc) {
        const float4* p = (const float4*)crow;
        const int b0 = (kc * 32 + h * 8) >> 2;
        float4 x0 = p[b0], x1 = p[b0 + 1], x2 = p[b0 + 4], x3 = p[b0 + 5];
        float f[16] = {x0.x,x0.y,x0.z,x0.w, x1.x,x1.y,x1.z,x1.w,
                       x2.x,x2.y,x2.z,x2.w, x3.x,x3.y,x3.z,x3.w};
        v16h hi, lo;
#pragma unroll
        for (int i = 0; i < 16; ++i) {
            _Float16 fh = (_Float16)f[i];
            hi[i] = fh;
            lo[i] = (_Float16)(f[i] - (float)fh);
        }
        bw[(size_t)(nt * 4 + kc * 2 + 0) * 32 + lane] = hi;
        bw[(size_t)(nt * 4 + kc * 2 + 1) * 32 + lane] = lo;
    }
    if (h == 0) {                 // exact f32 ||e_n||^2, once per column
        float s = 0.f;
#pragma unroll
        for (int k = 0; k < EMB_DIM; ++k) s += crow[k] * crow[k];
        en[nt * 16 + lr] = s;
    }
}

// Pass 1: fused split-f16 WMMA distance GEMM + row argmin + usage histogram.
// Inner loop: 4 pre-swizzled fragment loads + 1 enorm load + 6 WMMAs + min/select.
__global__ void __launch_bounds__(256)
vq_argmin(const float* __restrict__ z, const v16h* __restrict__ bw,
          const float* __restrict__ en, int* __restrict__ idx,
          unsigned* __restrict__ hist) {
    const int lane  = threadIdx.x & 31;
    const int wave  = threadIdx.x >> 5;
    const int half  = lane >> 4;
    const int lr    = lane & 15;
    const int rowBase = (blockIdx.x * 8 + wave) * 16;

    // loop-invariant A fragments for this wave's 16 z rows (K=0..31 and K=32..63)
    const float* zrow = z + (size_t)(rowBase + lr) * EMB_DIM;
    v16h a0h, a0l, a1h, a1l;
    load_a_frag(zrow, 0,  half, a0h, a0l);
    load_a_frag(zrow, 32, half, a1h, a1l);

    float minv[8];
    int   mini[8];
#pragma unroll
    for (int v = 0; v < 8; ++v) { minv[v] = 3.4e38f; mini[v] = 0; }

    for (int nt = 0; nt < NUM_EMB / 16; ++nt) {
        const v16h* bt = bw + (size_t)nt * 128 + lane;
        v16h b0h = bt[0], b0l = bt[32], b1h = bt[64], b1l = bt[96];
        const float enorm = en[nt * 16 + lr];

        // two independent accumulation chains (main + split correction)
        v8f c0 = {}, c1 = {};
        c0 = wmma_f16(a0h, b0h, c0);
        c1 = wmma_f16(a0l, b0h, c1);
        c0 = wmma_f16(a1h, b1h, c0);
        c1 = wmma_f16(a0h, b0l, c1);
        c1 = wmma_f16(a1l, b1h, c1);
        c1 = wmma_f16(a1h, b1l, c1);

        const int n = nt * 16 + lr;
#pragma unroll
        for (int v = 0; v < 8; ++v) {
            // argmin score: ||e||^2 - 2 z.e   (||z||^2 constant per row -> dropped)
            float s = enorm - 2.0f * (c0[v] + c1[v]);
            if (s < minv[v]) { minv[v] = s; mini[v] = n; }   // strict <: first-index ties
        }
    }

    // reduce min/argmin across the 16 lanes of each half-wave
#pragma unroll
    for (int mask = 1; mask < 16; mask <<= 1) {
#pragma unroll
        for (int v = 0; v < 8; ++v) {
            float ov = __shfl_xor(minv[v], mask, 32);
            int   oi = __shfl_xor(mini[v], mask, 32);
            if (ov < minv[v] || (ov == minv[v] && oi < mini[v])) { minv[v] = ov; mini[v] = oi; }
        }
    }

    if (lr == 0) {
#pragma unroll
        for (int v = 0; v < 8; ++v) {
            const int row = rowBase + half * 8 + v;   // C/D layout: m = v + 8*half
            idx[row] = mini[v];
            atomicAdd(&hist[mini[v]], 1u);
        }
    }
}

// Pass 2: gather codebook rows -> quantized_st output, fused SSE reduction,
// and indices written as float (harness output buffer is f32).
__global__ void __launch_bounds__(256)
vq_gather(const float* __restrict__ z, const float* __restrict__ cb,
          const int* __restrict__ idx, float* __restrict__ outQ,
          float* __restrict__ outIdx, float* __restrict__ sse) {
    const int t   = blockIdx.x * blockDim.x + threadIdx.x;  // N_ROWS*16 threads
    const int row = t >> 4;
    const int j   = t & 15;                                  // float4 slot within row
    const int k   = idx[row];

    const float4 q  = ((const float4*)(cb + (size_t)k   * EMB_DIM))[j];
    const float4 zz = ((const float4*)(z  + (size_t)row * EMB_DIM))[j];
    const float dx = q.x - zz.x, dy = q.y - zz.y, dz = q.z - zz.z, dw = q.w - zz.w;
    float s = dx * dx + dy * dy + dz * dz + dw * dw;

    ((float4*)outQ)[(size_t)row * 16 + j] = q;               // straight-through == quantized
    if (j == 0) outIdx[row] = (float)k;

#pragma unroll
    for (int m = 16; m; m >>= 1) s += __shfl_xor(s, m, 32);
    if ((threadIdx.x & 31) == 0) atomicAdd(sse, s);
}

// Pass 3: scalars — vq_loss = (1 + commitment) * MSE, perplexity from histogram.
__global__ void vq_finalize(const unsigned* __restrict__ hist, const float* __restrict__ sse,
                            float* __restrict__ outLoss, float* __restrict__ outPerp) {
    __shared__ float red[32];
    const int t = threadIdx.x;                // 1024 threads
    float p = (float)hist[t] / (float)N_ROWS;
    float h = p * logf(p + 1e-10f);
#pragma unroll
    for (int m = 16; m; m >>= 1) h += __shfl_xor(h, m, 32);
    if ((t & 31) == 0) red[t >> 5] = h;
    __syncthreads();
    if (t < 32) {
        float v = red[t];
#pragma unroll
        for (int m = 16; m; m >>= 1) v += __shfl_xor(v, m, 32);
        if (t == 0) {
            *outPerp = expf(-v);
            *outLoss = (1.0f + COMMIT) * (*sse) / (float)((size_t)N_ROWS * EMB_DIM);
        }
    }
}

extern "C" void kernel_launch(void* const* d_in, const int* in_sizes, int n_in,
                              void* d_out, int out_size, void* d_ws, size_t ws_size,
                              hipStream_t stream) {
    (void)in_sizes; (void)n_in; (void)out_size; (void)ws_size;

    const float* z  = (const float*)d_in[0];   // [16,2048,64] f32
    const float* cb = (const float*)d_in[1];   // [1024,64]    f32

    unsigned* hist = (unsigned*)d_ws;
    float*    sse  = (float*)((char*)d_ws + WS_SSE_OFF);
    int*      idx  = (int*)((char*)d_ws + WS_IDX_OFF);
    float*    en   = (float*)((char*)d_ws + WS_EN_OFF);
    v16h*     bw   = (v16h*)((char*)d_ws + WS_BF_OFF);

    float* out     = (float*)d_out;
    float* outQ    = out;                              // 2,097,152 elems
    float* outLoss = out + (size_t)N_ROWS * EMB_DIM;   // 1
    float* outIdx  = outLoss + 1;                      // 32,768
    float* outPerp = outIdx + N_ROWS;                  // 1

    vq_init    <<<1, 1024, 0, stream>>>(hist, sse);
    vq_prep    <<<8, 256, 0, stream>>>(cb, bw, en);
    vq_argmin  <<<N_ROWS / 128, 256, 0, stream>>>(z, bw, en, idx, hist);
    vq_gather  <<<(N_ROWS * 16) / 256, 256, 0, stream>>>(z, cb, idx, outQ, outIdx, sse);
    vq_finalize<<<1, 1024, 0, stream>>>(hist, sse, outLoss, outPerp);
}
